// MMDrbfLoss_33182917329043
// MI455X (gfx1250) — compile-verified, matrix operand order checked by reading, not run on hardware
//
#include <hip/hip_runtime.h>

typedef _Float16 half_t;
typedef __attribute__((ext_vector_type(16))) _Float16 v16h;
typedef __attribute__((ext_vector_type(8)))  _Float16 v8h;
typedef __attribute__((ext_vector_type(4)))  _Float16 v4h;
typedef __attribute__((ext_vector_type(8)))  float    v8f;
typedef __attribute__((ext_vector_type(4)))  float    v4f;

#define BS      4096
#define N_TOTAL 8192
#define DIM     1024

// ---------------------------------------------------------------- init ------
__global__ void init_kernel(float* scal) {
    if (threadIdx.x == 0) scal[0] = 0.0f;   // S = sum of row sq-norms
}

// ------------------------------------------------ pass 1: f16 copy + sq -----
// one block per row: convert row to f16, compute sq[row], atomically add to S
__global__ __launch_bounds__(256)
void prep_kernel(const float* __restrict__ src, const float* __restrict__ tgt,
                 half_t* __restrict__ Xh, float* __restrict__ sq,
                 float* __restrict__ scal) {
    const int row = blockIdx.x;
    const int t   = threadIdx.x;
    const float* p = (row < BS) ? (src + (size_t)row * DIM)
                                : (tgt + (size_t)(row - BS) * DIM);
    v4f x = *(const v4f*)(p + t * 4);
    v4h h;
    h.x = (half_t)x.x; h.y = (half_t)x.y; h.z = (half_t)x.z; h.w = (half_t)x.w;
    *(v4h*)(Xh + (size_t)row * DIM + t * 4) = h;
    float s = x.x * x.x + x.y * x.y + x.z * x.z + x.w * x.w;

    __shared__ float red[256];
    red[t] = s;
    __syncthreads();
    for (int off = 128; off > 0; off >>= 1) {
        if (t < off) red[t] += red[t + off];
        __syncthreads();
    }
    if (t == 0) {
        sq[row] = red[0];
        atomicAdd(&scal[0], red[0]);
    }
}

// --------------------------------------------- pass 2: column sums m[c] -----
__global__ __launch_bounds__(256)
void colsum_kernel(const float* __restrict__ src, const float* __restrict__ tgt,
                   float* __restrict__ m) {
    const int c = blockIdx.x * blockDim.x + threadIdx.x;
    if (c >= DIM) return;
    float s = 0.0f;
    for (int r = 0; r < BS; ++r) s += src[(size_t)r * DIM + c];
    for (int r = 0; r < BS; ++r) s += tgt[(size_t)r * DIM + c];
    m[c] = s;
}

// ------------------------------- pass 3: bandwidths from analytic sum(L2) ---
// sum(L2) = 2*n*S - 2*||m||^2  (no need to materialize the 8192x8192 matrix)
__global__ __launch_bounds__(256)
void bw_kernel(const float* __restrict__ m, float* __restrict__ scal) {
    const int t = threadIdx.x;
    float s = 0.0f;
    for (int c = t; c < DIM; c += 256) { float v = m[c]; s += v * v; }
    __shared__ float red[256];
    red[t] = s;
    __syncthreads();
    for (int off = 128; off > 0; off >>= 1) {
        if (t < off) red[t] += red[t + off];
        __syncthreads();
    }
    if (t == 0) {
        const float msq   = red[0];
        const float S     = scal[0];
        const float n     = (float)N_TOTAL;
        const float sumL2 = 2.0f * n * S - 2.0f * msq;
        float bw = sumL2 / (n * n - n);
        bw /= 4.0f;                       // KERNEL_MUL^(KERNEL_NUM//2) = 2^2
        for (int k = 0; k < 5; ++k) {     // store -1/bw_k, bw_k = bw * 2^k
            scal[2 + k] = -1.0f / bw;
            bw *= 2.0f;
        }
        scal[1] = 0.0f;                   // zero the global accumulator
    }
}

// ------------------------------------------------------- main WMMA kernel ---
// 16-bit A-fragment layout (ISA 7.12.2): lane L (<16) row M=L holds K 0..7 and
// 16..23; lane L+16 holds same row, K 8..15 and 24..31 -> two 16B loads/lane.
// For C = X*X^T the B-fragment (cols j) is rows j of X with the mirrored
// packing, so the same loader serves both operands.
__device__ __forceinline__ v16h load_frag(const half_t* __restrict__ Xh,
                                          int base_row, int k0, int lane) {
    const int row = base_row + (lane & 15);
    const int kh  = (lane >> 4) << 3;                 // 0 or 8
    const half_t* p = Xh + (size_t)row * DIM + k0 + kh;
    union { v16h v; v8h h[2]; } u;
    u.h[0] = *(const v8h*)p;                          // K = k0+kh .. +7
    u.h[1] = *(const v8h*)(p + 16);                   // K = k0+kh+16 .. +23
    return u.v;
}

__device__ __forceinline__ v8f wmma_f16(v16h a, v16h b, v8f c) {
    return __builtin_amdgcn_wmma_f32_16x16x32_f16(
        /*neg_a=*/false, a, /*neg_b=*/false, b,
        /*c_mod=*/(short)0, c, /*reuse_a=*/false, /*reuse_b=*/false);
}

__global__ __launch_bounds__(256)
void mmd_kernel(const half_t* __restrict__ Xh, const float* __restrict__ sq,
                const float* __restrict__ scal, float* __restrict__ acc_out) {
    const int lane = threadIdx.x & 31;
    const int wave = threadIdx.x >> 5;        // 8 waves
    const int wm   = wave & 3;                // 4 x 2 wave grid
    const int wn   = wave >> 2;
    const int bm   = blockIdx.x * 128;        // 128x128 block tile
    const int bn   = blockIdx.y * 128;
    const int rm   = bm + wm * 32;            // wave: 32 rows  (2 m-subtiles)
    const int rn   = bn + wn * 64;            // wave: 64 cols  (4 n-subtiles)

    v8f acc[2][4];
    const v8f vz = {};
    #pragma unroll
    for (int i = 0; i < 2; ++i)
        #pragma unroll
        for (int j = 0; j < 4; ++j) acc[i][j] = vz;

    for (int k0 = 0; k0 < DIM; k0 += 32) {
        v16h a0 = load_frag(Xh, rm,      k0, lane);
        v16h a1 = load_frag(Xh, rm + 16, k0, lane);
        v16h b0 = load_frag(Xh, rn,      k0, lane);
        v16h b1 = load_frag(Xh, rn + 16, k0, lane);
        v16h b2 = load_frag(Xh, rn + 32, k0, lane);
        v16h b3 = load_frag(Xh, rn + 48, k0, lane);
        acc[0][0] = wmma_f16(a0, b0, acc[0][0]);
        acc[0][1] = wmma_f16(a0, b1, acc[0][1]);
        acc[0][2] = wmma_f16(a0, b2, acc[0][2]);
        acc[0][3] = wmma_f16(a0, b3, acc[0][3]);
        acc[1][0] = wmma_f16(a1, b0, acc[1][0]);
        acc[1][1] = wmma_f16(a1, b1, acc[1][1]);
        acc[1][2] = wmma_f16(a1, b2, acc[1][2]);
        acc[1][3] = wmma_f16(a1, b3, acc[1][3]);
    }

    // fused epilogue: L2 -> 5-bandwidth gaussian sum -> signed reduction
    const float ni0 = scal[2], ni1 = scal[3], ni2 = scal[4],
                ni3 = scal[5], ni4 = scal[6];
    float lsum = 0.0f;
    #pragma unroll
    for (int mi = 0; mi < 2; ++mi) {
        const int rowb = rm + mi * 16 + ((lane >> 4) << 3);  // C/D layout
        #pragma unroll
        for (int nj = 0; nj < 4; ++nj) {
            const int col = rn + nj * 16 + (lane & 15);
            const float sqc = sq[col];
            #pragma unroll
            for (int e = 0; e < 8; ++e) {
                const float d2 = sq[rowb + e] + sqc - 2.0f * acc[mi][nj][e];
                lsum += __expf(d2 * ni0) + __expf(d2 * ni1) + __expf(d2 * ni2)
                      + __expf(d2 * ni3) + __expf(d2 * ni4);
            }
        }
    }
    // wave32 reduction
    #pragma unroll
    for (int off = 16; off > 0; off >>= 1) lsum += __shfl_xor(lsum, off, 32);

    __shared__ float wred[8];
    if (lane == 0) wred[wave] = lsum;
    __syncthreads();
    if (threadIdx.x == 0) {
        float bsum = 0.0f;
        #pragma unroll
        for (int w = 0; w < 8; ++w) bsum += wred[w];
        const float sign = ((bm < BS) == (bn < BS)) ? 1.0f : -1.0f;
        atomicAdd(acc_out, sign * bsum);
    }
}

// --------------------------------------------------------------- writeout ---
__global__ void writeout_kernel(const float* __restrict__ scal,
                                float* __restrict__ out) {
    out[0] = scal[1] / ((float)BS * (float)BS);
}

// ------------------------------------------------------------------ launch --
extern "C" void kernel_launch(void* const* d_in, const int* in_sizes, int n_in,
                              void* d_out, int out_size, void* d_ws, size_t ws_size,
                              hipStream_t stream) {
    const float* src = (const float*)d_in[0];
    const float* tgt = (const float*)d_in[1];

    char*   ws   = (char*)d_ws;
    half_t* Xh   = (half_t*)ws;                                   // 16 MB
    float*  sq   = (float*)(ws + (size_t)N_TOTAL * DIM * sizeof(half_t));
    float*  m    = sq + N_TOTAL;                                  // 1024 floats
    float*  scal = m + DIM;   // [0]=S  [1]=acc  [2..6]=-1/bw_k

    init_kernel<<<1, 1, 0, stream>>>(scal);
    prep_kernel<<<N_TOTAL, 256, 0, stream>>>(src, tgt, Xh, sq, scal);
    colsum_kernel<<<(DIM + 255) / 256, 256, 0, stream>>>(src, tgt, m);
    bw_kernel<<<1, 256, 0, stream>>>(m, scal);
    dim3 grid(N_TOTAL / 128, N_TOTAL / 128);                      // 64 x 64
    mmd_kernel<<<grid, 256, 0, stream>>>(Xh, sq, scal, scal + 1);
    writeout_kernel<<<1, 1, 0, stream>>>(scal, (float*)d_out);
}
